// Attention_78864189489362
// MI455X (gfx1250) — matrix operand order repaired and checked by reading, not compiled, not run on hardware
//
#include <hip/hip_runtime.h>
#include <hip/hip_bf16.h>
#include <stdint.h>

// ---------------------------------------------------------------------------
// MI455X (gfx1250) fused multi-head attention, bf16 WMMA, flash-style softmax
// Round 2: 64-key blocks in the attention loop (halves shuffle/fence overhead
// per element, doubles WMMA density), + global prefetch of next K/V block.
// ---------------------------------------------------------------------------

typedef __attribute__((ext_vector_type(16))) __bf16       v16bf;
typedef __attribute__((ext_vector_type(8)))  float        v8f;
typedef __attribute__((ext_vector_type(8)))  unsigned int v8u;

#define HEADS 4
#define CHEAD 32
#define CIN   256
#define CHID  128      // HEADS * CHEAD
#define NB    4
#define NSP   4096     // 64*64

// fp32 -> bf16 round-to-nearest-even
static __device__ __forceinline__ unsigned short f2bf(float f) {
  unsigned int u = __float_as_uint(f);
  unsigned int r = u + 0x7FFFu + ((u >> 16) & 1u);
  return (unsigned short)(r >> 16);
}

// A-fragment (16x32 bf16) K-offset for VGPR v (0..7), lane-half h.
// Per CDNA5 ISA 16-bit A layout: V0:K0,1 | V1:K2,3 | ... V4:K16,17 ...;
// lanes 16..31 use K+8. Pairs are contiguous -> coalesces to 2x b128/lane.
static __device__ __forceinline__ int kA(int v, int h) {
  return (v < 4 ? 2 * v : 16 + 2 * (v - 4)) + 8 * h;
}
// B-fragment (32x16 bf16) K-offset: same K order as kA (V0..3: K0..7,
// V4..7: K16..23; lanes 16..31 use K+8).
static __device__ __forceinline__ int kB(int v, int h) {
  return (v < 4 ? 2 * v : 16 + 2 * (v - 4)) + 8 * h;
}

static __device__ __forceinline__ v8f wmma_bf16(v8u a, v8u b, v8f c) {
  return __builtin_amdgcn_wmma_f32_16x16x32_bf16(
      false, __builtin_bit_cast(v16bf, a),
      false, __builtin_bit_cast(v16bf, b),
      (short)0, c, false, false);
}

// ---------------------------------------------------------------------------
// 1) convert x [b][c][n] fp32 -> xT [b][n][c] bf16
// ---------------------------------------------------------------------------
__global__ void cvt_x_kernel(const float* __restrict__ x,
                             unsigned short* __restrict__ xT) {
  int idx = blockIdx.x * blockDim.x + threadIdx.x;   // 4*256*4096 = 2^22
  int n = idx & (NSP - 1);
  int c = (idx >> 12) & (CIN - 1);
  int b = idx >> 20;
  xT[((size_t)b * NSP + n) * CIN + c] = f2bf(x[idx]);
}

// 2) convert weights to bf16 (row-major kept)
__global__ void cvt_w_kernel(const float* __restrict__ wqkv,
                             const float* __restrict__ wout,
                             unsigned short* __restrict__ wqb,
                             unsigned short* __restrict__ wob) {
  int i = blockIdx.x * blockDim.x + threadIdx.x;     // 98304 + 32768
  if (i < 3 * CHID * CIN) {
    wqb[i] = f2bf(wqkv[i]);
  } else {
    int j = i - 3 * CHID * CIN;
    if (j < CIN * CHID) wob[j] = f2bf(wout[j]);
  }
}

// ---------------------------------------------------------------------------
// 3) QKV projection: per batch [384,256] x [256,4096], one wave per 16x16 tile
//    Scatter into attention layouts: Q,K = [bh][n][32] (Q pre-scaled),
//    V = [bh][32][n].
// ---------------------------------------------------------------------------
__global__ void qkv_gemm_kernel(const unsigned short* __restrict__ wqb,
                                const unsigned short* __restrict__ xT,
                                unsigned short* __restrict__ Qb,
                                unsigned short* __restrict__ Kb,
                                unsigned short* __restrict__ Vb) {
  const int lane = threadIdx.x & 31;
  const int wave = threadIdx.x >> 5;
  const int tile = blockIdx.x * 8 + wave;      // 4 * 24 * 256 tiles
  const int b  = tile / (24 * 256);
  const int rem = tile % (24 * 256);
  const int mt = rem >> 8;                     // 0..23 (rows of w_qkv, /16)
  const int nt = rem & 255;                    // 0..255 (spatial, /16)
  const int half = lane >> 4;
  const int ml   = lane & 15;
  const int n = nt * 16 + ml;

  const unsigned short* arow = wqb + (size_t)(mt * 16 + ml) * CIN;
  const unsigned short* brow = xT + ((size_t)b * NSP + n) * CIN;

  v8f acc = {0.f, 0.f, 0.f, 0.f, 0.f, 0.f, 0.f, 0.f};
#pragma unroll
  for (int kk = 0; kk < CIN; kk += 32) {
    v8u au, bu;
#pragma unroll
    for (int v = 0; v < 8; ++v) {
      au[v] = *(const unsigned int*)(arow + kk + kA(v, half));
      bu[v] = *(const unsigned int*)(brow + kk + kB(v, half));
    }
    acc = wmma_bf16(au, bu, acc);
  }

  const float qscale = 0.1767766952966369f;    // 32^-0.5
#pragma unroll
  for (int r = 0; r < 8; ++r) {
    int o = mt * 16 + r + 8 * half;            // global out-channel (0..383)
    float val = acc[r];
    if (o < CHID) {                            // Q
      int h = o >> 5, c = o & 31;
      Qb[(((size_t)(b * HEADS + h)) * NSP + n) * CHEAD + c] = f2bf(val * qscale);
    } else if (o < 2 * CHID) {                 // K
      int o2 = o - CHID; int h = o2 >> 5, c = o2 & 31;
      Kb[(((size_t)(b * HEADS + h)) * NSP + n) * CHEAD + c] = f2bf(val);
    } else {                                   // V (channel-major)
      int o2 = o - 2 * CHID; int h = o2 >> 5, c = o2 & 31;
      Vb[(((size_t)(b * HEADS + h)) * CHEAD + c) * NSP + n] = f2bf(val);
    }
  }
}

// ---------------------------------------------------------------------------
// 4) Flash attention: one wave per (b,h, 16-query tile), 64-key blocks.
//    Online softmax with row stats replicated across the 16 lanes of each
//    C-row (butterfly shuffles amortized over 64 keys). P relayout
//    C-tile -> A-fragment through wave-private LDS (s_wait_dscnt fence).
// ---------------------------------------------------------------------------
__global__ void attn_kernel(const unsigned short* __restrict__ Qb,
                            const unsigned short* __restrict__ Kb,
                            const unsigned short* __restrict__ Vb,
                            unsigned short* __restrict__ aoT) {
  __shared__ unsigned short ldsP[8 * 16 * 64];     // 2KB per wave
  const int lane = threadIdx.x & 31;
  const int wave = threadIdx.x >> 5;
  const int task = blockIdx.x * 8 + wave;          // 16 * 256 tasks
  const int bh = task >> 8;                        // (b*4 + h)
  const int qt = task & 255;
  const int b = bh >> 2, h = bh & 3;
  const int half = lane >> 4;
  const int ml   = lane & 15;
  unsigned short* P = ldsP + wave * (16 * 64);

  const unsigned short* Kbase = Kb + (size_t)bh * NSP * CHEAD;
  const unsigned short* Vbase = Vb + (size_t)bh * CHEAD * NSP;

  // resident Q A-fragment: 16 queries x 32 channels
  v8u qa;
  {
    const unsigned short* qrow = Qb + (((size_t)bh) * NSP + qt * 16 + ml) * CHEAD;
#pragma unroll
    for (int v = 0; v < 8; ++v) qa[v] = *(const unsigned int*)(qrow + kA(v, half));
  }

  v8f o0 = {0.f, 0.f, 0.f, 0.f, 0.f, 0.f, 0.f, 0.f};
  v8f o1 = {0.f, 0.f, 0.f, 0.f, 0.f, 0.f, 0.f, 0.f};
  float mrow[8], lrow[8];
#pragma unroll
  for (int r = 0; r < 8; ++r) { mrow[r] = -1e30f; lrow[r] = 0.f; }

  for (int jb = 0; jb < NSP; jb += 64) {
    // prefetch next 64-key block of K and V into near caches
    {
      const unsigned short* kpf = Kbase + (size_t)(jb + 64 + ml * 4) * CHEAD;
      const unsigned short* vpf = Vbase + (size_t)(half * 16 + ml) * NSP + jb + 64;
      __builtin_prefetch(kpf, 0, 1);
      __builtin_prefetch(vpf, 0, 1);
    }

    // ---- S = Q K^T : four 16x16 tiles over this 64-key block ----
    v8f s[4];
#pragma unroll
    for (int t = 0; t < 4; ++t) {
      v8u kf;
      const unsigned short* krow = Kbase + (size_t)(jb + t * 16 + ml) * CHEAD;
#pragma unroll
      for (int v = 0; v < 8; ++v) kf[v] = *(const unsigned int*)(krow + kB(v, half));
      v8f z = {0.f, 0.f, 0.f, 0.f, 0.f, 0.f, 0.f, 0.f};
      s[t] = wmma_bf16(qa, kf, z);
    }

    // ---- online softmax (row = r + 8*half lives in 16 lanes of one VGPR) ----
#pragma unroll
    for (int r = 0; r < 8; ++r) {
      float sm = fmaxf(fmaxf(s[0][r], s[1][r]), fmaxf(s[2][r], s[3][r]));
      sm = fmaxf(sm, __shfl_xor(sm, 1, 32));
      sm = fmaxf(sm, __shfl_xor(sm, 2, 32));
      sm = fmaxf(sm, __shfl_xor(sm, 4, 32));
      sm = fmaxf(sm, __shfl_xor(sm, 8, 32));
      float mnew  = fmaxf(mrow[r], sm);
      float alpha = __expf(mrow[r] - mnew);
      mrow[r] = mnew;
      float p0 = __expf(s[0][r] - mnew);
      float p1 = __expf(s[1][r] - mnew);
      float p2 = __expf(s[2][r] - mnew);
      float p3 = __expf(s[3][r] - mnew);
      s[0][r] = p0; s[1][r] = p1; s[2][r] = p2; s[3][r] = p3;
      float rs = (p0 + p1) + (p2 + p3);
      rs += __shfl_xor(rs, 1, 32);
      rs += __shfl_xor(rs, 2, 32);
      rs += __shfl_xor(rs, 4, 32);
      rs += __shfl_xor(rs, 8, 32);
      lrow[r] = lrow[r] * alpha + rs;
      o0[r] *= alpha;
      o1[r] *= alpha;
    }

    // ---- P: C-tiles -> A-fragments through wave-private LDS (16x64 bf16) ----
#pragma unroll
    for (int r = 0; r < 8; ++r) {
      int i = r + 8 * half;
#pragma unroll
      for (int t = 0; t < 4; ++t)
        P[i * 64 + t * 16 + ml] = f2bf(s[t][r]);
    }
    asm volatile("s_wait_dscnt 0" ::: "memory");   // wave-local LDS fence
    v8u pa0, pa1;
#pragma unroll
    for (int v = 0; v < 8; ++v) {
      pa0[v] = *(const unsigned int*)(P + ml * 64 + kA(v, half));
      pa1[v] = *(const unsigned int*)(P + ml * 64 + 32 + kA(v, half));
    }

    // ---- O += P V : two 16-channel halves x two 32-key sub-blocks ----
#pragma unroll
    for (int t = 0; t < 2; ++t) {                  // channel half
      const unsigned short* vrow = Vbase + (size_t)(t * 16 + ml) * NSP + jb;
      v8u vf0, vf1;
#pragma unroll
      for (int v = 0; v < 8; ++v) {
        vf0[v] = *(const unsigned int*)(vrow + kB(v, half));
        vf1[v] = *(const unsigned int*)(vrow + 32 + kB(v, half));
      }
      if (t == 0) { o0 = wmma_bf16(pa0, vf0, o0); o0 = wmma_bf16(pa1, vf1, o0); }
      else        { o1 = wmma_bf16(pa0, vf0, o1); o1 = wmma_bf16(pa1, vf1, o1); }
    }
  }

  // ---- normalize, write aoT [b][n][h*32+c] bf16 for the out-proj GEMM ----
#pragma unroll
  for (int r = 0; r < 8; ++r) {
    int n = qt * 16 + r + 8 * half;
    float inv = 1.0f / lrow[r];
    size_t base = ((size_t)b * NSP + n) * CHID + h * CHEAD;
    aoT[base + ml]      = f2bf(o0[r] * inv);
    aoT[base + 16 + ml] = f2bf(o1[r] * inv);
  }
}

// ---------------------------------------------------------------------------
// 5) Output projection: [256,128] x [128,4096] per batch + bias, fp32 out
// ---------------------------------------------------------------------------
__global__ void out_proj_kernel(const unsigned short* __restrict__ wob,
                                const unsigned short* __restrict__ aoT,
                                const float* __restrict__ bias,
                                float* __restrict__ out) {
  const int lane = threadIdx.x & 31;
  const int wave = threadIdx.x >> 5;
  const int tile = blockIdx.x * 8 + wave;     // 4 * 16 * 256 tiles
  const int b   = tile >> 12;                 // 16*256 = 4096 tiles per batch
  const int rem = tile & 4095;
  const int mt = rem >> 8;                    // 0..15
  const int nt = rem & 255;                   // 0..255
  const int half = lane >> 4;
  const int ml   = lane & 15;

  const unsigned short* arow = wob + (size_t)(mt * 16 + ml) * CHID;
  const unsigned short* brow = aoT + ((size_t)b * NSP + nt * 16 + ml) * CHID;

  v8f acc = {0.f, 0.f, 0.f, 0.f, 0.f, 0.f, 0.f, 0.f};
#pragma unroll
  for (int kk = 0; kk < CHID; kk += 32) {
    v8u au, bu;
#pragma unroll
    for (int v = 0; v < 8; ++v) {
      au[v] = *(const unsigned int*)(arow + kk + kA(v, half));
      bu[v] = *(const unsigned int*)(brow + kk + kB(v, half));
    }
    acc = wmma_bf16(au, bu, acc);
  }
#pragma unroll
  for (int r = 0; r < 8; ++r) {
    int o = mt * 16 + r + 8 * half;
    int n = nt * 16 + ml;
    out[((size_t)(b * CIN) + o) * NSP + n] = acc[r] + bias[o];
  }
}

// ---------------------------------------------------------------------------
extern "C" void kernel_launch(void* const* d_in, const int* in_sizes, int n_in,
                              void* d_out, int out_size, void* d_ws, size_t ws_size,
                              hipStream_t stream) {
  (void)in_sizes; (void)n_in; (void)out_size; (void)ws_size;
  const float* x    = (const float*)d_in[0];   // [4,256,64,64]
  const float* wqkv = (const float*)d_in[1];   // [384,256]
  const float* wout = (const float*)d_in[2];   // [256,128]
  const float* bout = (const float*)d_in[3];   // [256]
  float* out = (float*)d_out;                  // [4,256,64,64]

  char* ws = (char*)d_ws;
  unsigned short* xT  = (unsigned short*)(ws);              // 8,388,608 B
  unsigned short* wqb = (unsigned short*)(ws +  8388608);   //   196,608 B
  unsigned short* wob = (unsigned short*)(ws +  8585216);   //    65,536 B
  unsigned short* Qb  = (unsigned short*)(ws +  8650752);   // 4,194,304 B
  unsigned short* Kb  = (unsigned short*)(ws + 12845056);   // 4,194,304 B
  unsigned short* Vb  = (unsigned short*)(ws + 17039360);   // 4,194,304 B
  unsigned short* aoT = (unsigned short*)(ws + 21233664);   // 4,194,304 B

  cvt_x_kernel   <<<16384, 256, 0, stream>>>(x, xT);
  cvt_w_kernel   <<<  512, 256, 0, stream>>>(wqkv, wout, wqb, wob);
  qkv_gemm_kernel<<< 3072, 256, 0, stream>>>(wqb, xT, Qb, Kb, Vb);
  attn_kernel    <<<  512, 256, 0, stream>>>(Qb, Kb, Vb, aoT);
  out_proj_kernel<<< 2048, 256, 0, stream>>>(wob, aoT, bout, out);
}
